// GNNAttentionResidualModel_73632919323004
// MI455X (gfx1250) — matrix operand order repaired and checked by reading, not compiled, not run on hardware
//
#include <hip/hip_runtime.h>
#include <hip/hip_bf16.h>
#include <math.h>

#define HCN 128
#define BN_EPS 1e-5f
#define NEG_SLOPE 0.2f

typedef __attribute__((ext_vector_type(16))) _Float16 v16h;
typedef __attribute__((ext_vector_type(8)))  _Float16 v8h;
typedef __attribute__((ext_vector_type(8)))  float    v8f;

// ---------------- elementwise helpers ----------------

__global__ void cvt_f16_k(const float* __restrict__ a, _Float16* __restrict__ o, int n) {
  int t = blockIdx.x * blockDim.x + threadIdx.x;
  if (t < n) o[t] = (_Float16)a[t];
}

__global__ void fillf_k(float* __restrict__ p, float v, int n) {
  int t = blockIdx.x * blockDim.x + threadIdx.x;
  if (t < n) p[t] = v;
}

__global__ void bias_init_k(float* __restrict__ out, const float* __restrict__ bias, int n128) {
  int t = blockIdx.x * blockDim.x + threadIdx.x;
  if (t < n128) out[t] = bias[t & (HCN - 1)];
}

__global__ void hcat_f16_k(const float* __restrict__ hf, const float* __restrict__ hb,
                           _Float16* __restrict__ o, int n256) {
  int t = blockIdx.x * blockDim.x + threadIdx.x;
  if (t >= n256) return;
  int n = t >> 8, c = t & 255;
  float v = (c < HCN) ? hf[n * HCN + c] : hb[n * HCN + (c - HCN)];
  o[t] = (_Float16)v;
}

// ---------------- WMMA GEMM (A row-major f16, B pre-packed, C[N,128] f32) ----------------
// B packed layout: [K/32][nt:8][lane:32][e:16] halves where
//   col = nt*16 + (lane&15), K = kt*32 + (lane>>4)*8 + (e&7) + (e>>3)*16
__global__ void pack_b_k(const float* __restrict__ W, _Float16* __restrict__ Bp, int K) {
  int t = blockIdx.x * blockDim.x + threadIdx.x;
  int total = (K >> 5) * 8 * 32;
  if (t >= total) return;
  int lane = t & 31;
  int nt = (t >> 5) & 7;
  int kt = t >> 8;
  int col = nt * 16 + (lane & 15);
  int k0 = kt * 32 + (lane >> 4) * 8;
  _Float16* dst = Bp + (size_t)t * 16;
#pragma unroll
  for (int e = 0; e < 16; ++e) dst[e] = (_Float16)W[(size_t)(k0 + (e & 7) + (e >> 3) * 16) * HCN + col];
}

__global__ __launch_bounds__(128)
void gemm_wmma_n128_k(const _Float16* __restrict__ A, const _Float16* __restrict__ Bp,
                      float* __restrict__ C, const float* __restrict__ bias,
                      int K, int relu, int M) {
  const int lane = threadIdx.x & 31;
  const int wave = blockIdx.x * (blockDim.x >> 5) + (threadIdx.x >> 5);
  const int mt = wave;
  if (mt * 16 >= M) return;
  const int lrow = mt * 16 + (lane & 15);
  const int kh = lane >> 4;       // which K-half this lane supplies for A
  const int KT = K >> 5;

  v8f acc[8] = {};
  const _Float16* arow = A + (size_t)lrow * K + kh * 8;
  const _Float16* bln  = Bp + (size_t)lane * 16;

  for (int kt = 0; kt < KT; ++kt) {
    if (kt + 1 < KT) __builtin_prefetch(arow + (kt + 1) * 32, 0, 0);
    v8h alo = *(const v8h*)(arow + kt * 32);        // K = kt*32 + kh*8 .. +7
    v8h ahi = *(const v8h*)(arow + kt * 32 + 16);   // K = kt*32 + 16 + kh*8 .. +7
    v16h a;
#pragma unroll
    for (int e = 0; e < 8; ++e) { a[e] = alo[e]; a[8 + e] = ahi[e]; }
    const _Float16* bbase = bln + (size_t)kt * (8 * 32 * 16);
#pragma unroll
    for (int nt = 0; nt < 8; ++nt) {
      v16h b = *(const v16h*)(bbase + (size_t)nt * (32 * 16));
      acc[nt] = __builtin_amdgcn_wmma_f32_16x16x32_f16(
          false, a, false, b, (short)0, acc[nt], false, false);
    }
  }

  // C/D layout: VGPR r -> row = mt*16 + (lane>=16 ? 8 : 0) + r; col = nt*16 + (lane&15)
  const int orow0 = mt * 16 + kh * 8;
  const int colb = lane & 15;
#pragma unroll
  for (int nt = 0; nt < 8; ++nt) {
    int col = nt * 16 + colb;
    float bv = bias ? bias[col] : 0.0f;
#pragma unroll
    for (int r = 0; r < 8; ++r) {
      float v = acc[nt][r] + bv;
      if (relu) v = v > 0.0f ? v : 0.0f;
      C[(size_t)(orow0 + r) * HCN + col] = v;
    }
  }
}

// ---------------- attention terms ----------------
__global__ void attn_k(const float* __restrict__ hg, const float* __restrict__ asrc,
                       const float* __restrict__ adst, float* __restrict__ a_s,
                       float* __restrict__ a_d, int N) {
  int t = blockIdx.x * blockDim.x + threadIdx.x;
  if (t >= N * 2) return;
  int n = t >> 1, h = t & 1;
  const float* hp = hg + (size_t)n * HCN + h * 64;
  const float* ps = asrc + h * 64;
  const float* pd = adst + h * 64;
  float ss = 0.f, sd = 0.f;
#pragma unroll 4
  for (int c = 0; c < 64; ++c) { float v = hp[c]; ss += v * ps[c]; sd += v * pd[c]; }
  a_s[t] = ss; a_d[t] = sd;
}

// ---------------- edge softmax passes ----------------
__device__ inline void atomicMaxF(float* addr, float v) {
  if (v >= 0.0f) atomicMax((int*)addr, __float_as_int(v));
  else           atomicMin((unsigned int*)addr, __float_as_uint(v));
}

__global__ void edge_max_k(const int* __restrict__ ei, int E, int N,
                           const float* __restrict__ a_s, const float* __restrict__ a_d,
                           float* __restrict__ emax) {
  int i = blockIdx.x * blockDim.x + threadIdx.x;
  int tot = E + N;
  if (i >= tot) return;
  int s, d;
  if (i < E) { s = ei[i]; d = ei[E + i]; } else { s = i - E; d = s; }
#pragma unroll
  for (int h = 0; h < 2; ++h) {
    float e = a_s[s * 2 + h] + a_d[d * 2 + h];
    e = e > 0.f ? e : NEG_SLOPE * e;
    atomicMaxF(&emax[d * 2 + h], e);
  }
}

__global__ void edge_exp_k(const int* __restrict__ ei, int E, int N,
                           const float* __restrict__ a_s, const float* __restrict__ a_d,
                           const float* __restrict__ emax, float* __restrict__ denom,
                           float* __restrict__ pbuf) {
  int i = blockIdx.x * blockDim.x + threadIdx.x;
  int tot = E + N;
  if (i >= tot) return;
  int s, d;
  if (i < E) { s = ei[i]; d = ei[E + i]; } else { s = i - E; d = s; }
#pragma unroll
  for (int h = 0; h < 2; ++h) {
    float e = a_s[s * 2 + h] + a_d[d * 2 + h];
    e = e > 0.f ? e : NEG_SLOPE * e;
    float p = __expf(e - emax[d * 2 + h]);
    pbuf[(size_t)i * 2 + h] = p;
    atomicAdd(&denom[d * 2 + h], p);
  }
}

// wave per edge; each lane handles 4 contiguous channels (float4 gather + 4 atomic adds)
__global__ void edge_acc_k(const int* __restrict__ ei, int E, int N,
                           const float* __restrict__ hg, const float* __restrict__ pbuf,
                           const float* __restrict__ denom, float* __restrict__ out) {
  int t = blockIdx.x * blockDim.x + threadIdx.x;
  int edge = t >> 5;
  int tot = E + N;
  if (edge >= tot) return;
  int lane = t & 31;
  int s, d;
  if (edge < E) { s = ei[edge]; d = ei[E + edge]; } else { s = edge - E; d = s; }
  int c0 = lane * 4;
  int h = c0 >> 6;
  float alpha = pbuf[(size_t)edge * 2 + h] / (denom[d * 2 + h] + 1e-16f);
  float4 hv = *(const float4*)(hg + (size_t)s * HCN + c0);
  float* op = out + (size_t)d * HCN + c0;
  atomicAdd(op + 0, hv.x * alpha);
  atomicAdd(op + 1, hv.y * alpha);
  atomicAdd(op + 2, hv.z * alpha);
  atomicAdd(op + 3, hv.w * alpha);
}

// ---------------- batch norm ----------------
__global__ void bn_reduce_k(const float* __restrict__ m, float* __restrict__ gs,
                            float* __restrict__ gq, int N) {
  int c = threadIdx.x;  // blockDim == 128
  float s = 0.f, q = 0.f;
  for (int r = blockIdx.x; r < N; r += gridDim.x) {
    float v = m[(size_t)r * HCN + c];
    s += v; q += v * v;
  }
  atomicAdd(&gs[c], s);
  atomicAdd(&gq[c], q);
}

__global__ void bn_fin_k(const float* __restrict__ m, const float* __restrict__ gs,
                         const float* __restrict__ gq, const float* __restrict__ g,
                         const float* __restrict__ b, const float* __restrict__ resid,
                         float* __restrict__ out, int N) {
  int t = blockIdx.x * blockDim.x + threadIdx.x;
  if (t >= N * HCN) return;
  int c = t & (HCN - 1);
  float invN = 1.0f / (float)N;
  float mu = gs[c] * invN;
  float var = gq[c] * invN - mu * mu;
  out[t] = (m[t] - mu) * rsqrtf(var + BN_EPS) * g[c] + b[c] + resid[t];
}

// ---------------- host orchestration ----------------
extern "C" void kernel_launch(void* const* d_in, const int* in_sizes, int n_in,
                              void* d_out, int out_size, void* d_ws, size_t ws_size,
                              hipStream_t stream) {
  const float* x       = (const float*)d_in[0];
  const int*   fwd_ei  = (const int*)d_in[2];
  const int*   bwd_ei  = (const int*)d_in[3];
  const float* fwd0_w    = (const float*)d_in[4];
  const float* fwd0_asrc = (const float*)d_in[5];
  const float* fwd0_adst = (const float*)d_in[6];
  const float* fwd0_b    = (const float*)d_in[7];
  const float* bwd0_w    = (const float*)d_in[8];
  const float* bwd0_asrc = (const float*)d_in[9];
  const float* bwd0_adst = (const float*)d_in[10];
  const float* bwd0_b    = (const float*)d_in[11];
  const float* fwd1_w    = (const float*)d_in[12];
  const float* fwd1_asrc = (const float*)d_in[13];
  const float* fwd1_adst = (const float*)d_in[14];
  const float* fwd1_b    = (const float*)d_in[15];
  const float* bwd1_w    = (const float*)d_in[16];
  const float* bwd1_asrc = (const float*)d_in[17];
  const float* bwd1_adst = (const float*)d_in[18];
  const float* bwd1_b    = (const float*)d_in[19];
  const float* merge0_w  = (const float*)d_in[20];
  const float* merge0_b  = (const float*)d_in[21];
  const float* bn0_g     = (const float*)d_in[22];
  const float* bn0_b     = (const float*)d_in[23];
  const float* merge1_w  = (const float*)d_in[24];
  const float* merge1_b  = (const float*)d_in[25];
  const float* bn1_g     = (const float*)d_in[26];
  const float* bn1_b     = (const float*)d_in[27];
  const float* res0_w    = (const float*)d_in[28];
  const float* res0_b    = (const float*)d_in[29];

  const int N = in_sizes[0] / 256;   // 32768
  const int E = in_sizes[2] / 2;     // 524288
  const int TOT = E + N;

  // ---- workspace carve-up ----
  char* w = (char*)d_ws;
  size_t off = 0;
  auto carve = [&](size_t bytes) -> void* {
    void* p = w + off;
    off += (bytes + 255) & ~(size_t)255;
    return p;
  };
  _Float16* A16  = (_Float16*)carve((size_t)N * 256 * 2);   // f16 layer input / concat
  _Float16* Wp   = (_Float16*)carve((size_t)256 * 128 * 2); // packed weights (reused)
  float* hgf  = (float*)carve((size_t)N * HCN * 4);  // fwd h = x@W ; later merge output m
  float* hgb  = (float*)carve((size_t)N * HCN * 4);  // bwd h = x@W
  float* hf   = (float*)carve((size_t)N * HCN * 4);  // fwd GAT aggregation out
  float* hb   = (float*)carve((size_t)N * HCN * 4);  // bwd GAT aggregation out
  float* resb = (float*)carve((size_t)N * HCN * 4);  // residual / layer-0 output h0
  float* a_s  = (float*)carve((size_t)N * 2 * 4);
  float* a_d  = (float*)carve((size_t)N * 2 * 4);
  float* emax = (float*)carve((size_t)N * 2 * 4);
  float* denm = (float*)carve((size_t)N * 2 * 4);
  float* pbuf = (float*)carve((size_t)TOT * 2 * 4);
  float* bnsum = (float*)carve(HCN * 4);
  float* bnsq  = (float*)carve(HCN * 4);
  (void)ws_size; (void)n_in; (void)out_size;

  const int TB = 256;
  auto blk = [&](long n) { return dim3((unsigned)((n + TB - 1) / TB)); };

  auto gemm = [&](const _Float16* A, const float* W, int K, float* C,
                  const float* bias, int relu) {
    int pt = (K >> 5) * 8 * 32;
    pack_b_k<<<blk(pt), dim3(TB), 0, stream>>>(W, Wp, K);
    int waves = N / 16;
    gemm_wmma_n128_k<<<dim3((unsigned)((waves + 3) / 4)), dim3(128), 0, stream>>>(
        A, Wp, C, bias, K, relu, N);
  };

  auto gat_edges = [&](const float* hg, const int* ei, const float* asrc,
                       const float* adst, const float* bias, float* out) {
    attn_k<<<blk((long)N * 2), dim3(TB), 0, stream>>>(hg, asrc, adst, a_s, a_d, N);
    fillf_k<<<blk((long)N * 2), dim3(TB), 0, stream>>>(emax, -INFINITY, N * 2);
    fillf_k<<<blk((long)N * 2), dim3(TB), 0, stream>>>(denm, 0.0f, N * 2);
    bias_init_k<<<blk((long)N * HCN), dim3(TB), 0, stream>>>(out, bias, N * HCN);
    edge_max_k<<<blk(TOT), dim3(TB), 0, stream>>>(ei, E, N, a_s, a_d, emax);
    edge_exp_k<<<blk(TOT), dim3(TB), 0, stream>>>(ei, E, N, a_s, a_d, emax, denm, pbuf);
    edge_acc_k<<<blk((long)TOT * 32), dim3(TB), 0, stream>>>(ei, E, N, hg, pbuf, denm, out);
  };

  auto batchnorm_residual = [&](const float* m, const float* g, const float* b,
                                const float* resid, float* out) {
    fillf_k<<<dim3(1), dim3(HCN), 0, stream>>>(bnsum, 0.0f, HCN);
    fillf_k<<<dim3(1), dim3(HCN), 0, stream>>>(bnsq, 0.0f, HCN);
    bn_reduce_k<<<dim3(256), dim3(HCN), 0, stream>>>(m, bnsum, bnsq, N);
    bn_fin_k<<<blk((long)N * HCN), dim3(TB), 0, stream>>>(m, bnsum, bnsq, g, b, resid, out, N);
  };

  // ================= layer 0 (256 -> 128) =================
  cvt_f16_k<<<blk((long)N * 256), dim3(TB), 0, stream>>>(x, A16, N * 256);
  gemm(A16, fwd0_w, 256, hgf, nullptr, 0);
  gemm(A16, bwd0_w, 256, hgb, nullptr, 0);
  gemm(A16, res0_w, 256, resb, res0_b, 0);           // residual transform
  gat_edges(hgf, fwd_ei, fwd0_asrc, fwd0_adst, fwd0_b, hf);
  gat_edges(hgb, bwd_ei, bwd0_asrc, bwd0_adst, bwd0_b, hb);
  hcat_f16_k<<<blk((long)N * 256), dim3(TB), 0, stream>>>(hf, hb, A16, N * 256);
  gemm(A16, merge0_w, 256, hgf /* m */, merge0_b, 1);  // relu(concat @ W + b)
  batchnorm_residual(hgf, bn0_g, bn0_b, resb, resb);   // resb <- h0

  // ================= layer 1 (128 -> 128, identity residual) =================
  cvt_f16_k<<<blk((long)N * 128), dim3(TB), 0, stream>>>(resb, A16, N * 128);
  gemm(A16, fwd1_w, 128, hgf, nullptr, 0);
  gemm(A16, bwd1_w, 128, hgb, nullptr, 0);
  gat_edges(hgf, fwd_ei, fwd1_asrc, fwd1_adst, fwd1_b, hf);
  gat_edges(hgb, bwd_ei, bwd1_asrc, bwd1_adst, bwd1_b, hb);
  hcat_f16_k<<<blk((long)N * 256), dim3(TB), 0, stream>>>(hf, hb, A16, N * 256);
  gemm(A16, merge1_w, 256, hgf /* m */, merge1_b, 1);
  batchnorm_residual(hgf, bn1_g, bn1_b, resb, (float*)d_out);
}
